// LSTMNet_21526376088357
// MI455X (gfx1250) — matrix operand order repaired and checked by reading, not compiled, not run on hardware
//
#include <hip/hip_runtime.h>
#include <hip/hip_bf16.h>

// CDNA5 / gfx1250 WMMA vector types
typedef __attribute__((ext_vector_type(16))) __bf16 v16bf;
typedef __attribute__((ext_vector_type(8)))  float  v8f;

#define B_TOTAL 256
#define T_LEN   2048

__device__ __forceinline__ float fast_sigmoid(float x) {
    return 1.0f / (1.0f + __expf(-x));
}
__device__ __forceinline__ float fast_tanh(float x) {
    return 2.0f / (1.0f + __expf(-2.0f * x)) - 1.0f;
}

// One workgroup handles a block of 16 batch rows for the whole T-step
// recurrence. Weights are converted to bf16 and swizzled into LDS once;
// the B-fragments of the last RES_KS k-steps (covering all of Whh) are
// then hoisted into VGPRs for the entire time loop, removing LDS loads
// (and their s_wait_dscnt stalls) from the recurrent critical path.
// c/h state lives in VGPRs; h_t is exchanged through LDS each step.
//   IN  = input feature dim, H = hidden dim, K = IN + H (mult of 32)
//   F32IN: input activations are fp32 (layer 1) vs bf16 (layers 2/3)
template<int IN, int H, bool F32IN, bool WRITE_Y, bool WRITE_FINAL>
__global__ __launch_bounds__(256, 1)
void lstm_layer_kernel(const float*  __restrict__ xf,   // [B,T,IN] fp32 (or null)
                       const __bf16* __restrict__ xb,   // [B,T,IN] bf16 (or null)
                       const int*    __restrict__ lengths,
                       const float*  __restrict__ Wih,  // [4H, IN]
                       const float*  __restrict__ Whh,  // [4H, H]
                       const float*  __restrict__ bih,  // [4H]
                       const float*  __restrict__ bhh,  // [4H]
                       __bf16*       __restrict__ y_out,   // [B,T,H] bf16
                       float*        __restrict__ h_final) // [B,H] fp32
{
    constexpr int K  = IN + H;       // fused GEMM depth
    constexpr int KS = K / 32;       // k-steps of 32 (bf16 WMMA)
    constexpr int NW = H / 16;       // waves per block == cell tiles
    constexpr int NT = 4 * NW;       // gate tiles (i,f,g,o)
    // register-resident k-steps (from the end => entire Whh part), capped at
    // 4*4 = 16 fragments = 128 VGPRs to stay under the 256-VGPR direct window
    constexpr int RES_KS = (KS <= 4) ? KS : 4;
    constexpr int LDS_KS = KS - RES_KS;   // k-steps still read from LDS

    extern __shared__ __align__(32) char smem[];
    __bf16* Wlds = (__bf16*)smem;                                   // NT*KS*32*16 bf16
    __bf16* Alds = (__bf16*)(smem + (size_t)NT * KS * 512 * sizeof(__bf16)); // 16*K bf16

    const int tid  = threadIdx.x;
    const int lane = tid & 31;
    const int wid  = tid >> 5;
    const int mb   = blockIdx.x;      // which block of 16 batch rows
    const int m16  = lane & 15;
    const int kh   = lane >> 4;       // half-lane selector (layout)
    const int n0   = wid * 16;        // this wave's hidden-column base

    // ---- one-time: convert fp32 weights -> bf16, swizzled into this code's
    //      WMMA B-fragment order (contiguous k, matching A; B 32x16: lane -> N=lane%16,
    //      k-half=lane/16; elem j -> K = 16*kh + j)
    for (int idx = tid; idx < NT * KS * 512; idx += blockDim.x) {
        int j  = idx & 15;
        int l  = (idx >> 4) & 31;
        int ks = (idx >> 9) % KS;
        int tn = idx / (KS * 512);
        int n  = tn * 16 + (l & 15);
        int k  = ks * 32 + (l >> 4) * 16 + j;
        float w = (k < IN) ? Wih[n * IN + k] : Whh[n * H + (k - IN)];
        Wlds[idx] = (__bf16)w;
    }

    // Per-lane gate biases (column-dependent only) and sequence lengths
    float biasg[4];
#pragma unroll
    for (int g = 0; g < 4; ++g) {
        int col = g * H + n0 + m16;
        biasg[g] = bih[col] + bhh[col];
    }
    int lenr[8];
#pragma unroll
    for (int r = 0; r < 8; ++r)
        lenr[r] = lengths[mb * 16 + kh * 8 + r];

    __syncthreads();   // Wlds fully written

    // ---- hoist the recurrent-weight B fragments into registers for all T
    v16bf bres[4][RES_KS];
#pragma unroll
    for (int g = 0; g < 4; ++g)
#pragma unroll
        for (int rk = 0; rk < RES_KS; ++rk) {
            int tn = g * NW + wid;
            int ks = LDS_KS + rk;
            bres[g][rk] = *(const v16bf*)
                &Wlds[((size_t)(tn * KS + ks) * 32 + lane) * 16];
        }

    v8f c_state = {};   // cell state, fp32, tile element (row kh*8+r, col n0+m16)
    v8f h_state = {};   // hidden state, fp32

    for (int t = 0; t < T_LEN; ++t) {
        // ---- stage A = [x_t | h_{t-1}] as bf16 row-major [16][K] in LDS
        for (int idx = tid; idx < 16 * IN; idx += blockDim.x) {
            int m = idx / IN, k = idx % IN;
            long long base = ((long long)(mb * 16 + m) * T_LEN + t) * IN + k;
            float v = F32IN ? xf[base] : (float)xb[base];
            Alds[m * K + k] = (__bf16)v;
        }
#pragma unroll
        for (int r = 0; r < 8; ++r) {
            int row = kh * 8 + r;
            Alds[row * K + IN + n0 + m16] = (__bf16)h_state[r];
        }
        // prefetch next timestep's activation rows (global_prefetch_b8)
        if (t + 1 < T_LEN && lane == 0) {
            long long base = ((long long)(mb * 16 + wid) * T_LEN + (t + 1)) * IN;
            if (F32IN) __builtin_prefetch(xf + base, 0, 1);
            else       __builtin_prefetch(xb + base, 0, 1);
        }
        __syncthreads();

        // ---- gates = A @ [Wih|Whh]^T + bias, 4 gate tiles per wave
        v8f acc[4];
#pragma unroll
        for (int g = 0; g < 4; ++g)
#pragma unroll
            for (int r = 0; r < 8; ++r)
                acc[g][r] = biasg[g];

#pragma unroll
        for (int ks = 0; ks < KS; ++ks) {
            // A fragment (contiguous k order, same as B; 16-bit A 16x32): lane m=lane%16,
            // VGPR v holds K = 16*kh + 2*v + {0,1}
            v16bf a;
#pragma unroll
            for (int v = 0; v < 8; ++v) {
                int k0 = ks * 32 + kh * 16 + 2 * v;
                a[2 * v]     = Alds[m16 * K + k0];
                a[2 * v + 1] = Alds[m16 * K + k0 + 1];
            }
#pragma unroll
            for (int g = 0; g < 4; ++g) {
                if (ks >= LDS_KS) {
                    acc[g] = __builtin_amdgcn_wmma_f32_16x16x32_bf16(
                        false, a, false, bres[g][ks - LDS_KS], (short)0,
                        acc[g], false, false);
                } else {
                    int tn = g * NW + wid;
                    const v16bf* bp = (const v16bf*)
                        &Wlds[((size_t)(tn * KS + ks) * 32 + lane) * 16];
                    acc[g] = __builtin_amdgcn_wmma_f32_16x16x32_bf16(
                        false, a, false, *bp, (short)0, acc[g], false, false);
                }
            }
        }

        // ---- elementwise LSTM cell + packed-sequence masking
#pragma unroll
        for (int r = 0; r < 8; ++r) {
            float iv = fast_sigmoid(acc[0][r]);
            float fv = fast_sigmoid(acc[1][r]);
            float gv = fast_tanh(acc[2][r]);
            float ov = fast_sigmoid(acc[3][r]);
            float cn = fv * c_state[r] + iv * gv;
            float hn = ov * fast_tanh(cn);
            bool valid = t < lenr[r];
            c_state[r] = valid ? cn : c_state[r];
            h_state[r] = valid ? hn : h_state[r];
            if (WRITE_Y) {
                long long o =
                    ((long long)(mb * 16 + kh * 8 + r) * T_LEN + t) * H + n0 + m16;
                y_out[o] = (__bf16)h_state[r];
            }
        }
        __syncthreads();   // protect Alds before next step's staging writes
    }

    if (WRITE_FINAL) {
#pragma unroll
        for (int r = 0; r < 8; ++r)
            h_final[(mb * 16 + kh * 8 + r) * H + n0 + m16] = h_state[r];
    }
}

// Tiny MLP head: relu(h@fc1^T+b1) -> relu(@fc2^T+b2) -> @fco^T+bo
__global__ void head_kernel(const float* __restrict__ h3,   // [256,32]
                            const float* __restrict__ fc1w, const float* __restrict__ fc1b,
                            const float* __restrict__ fc2w, const float* __restrict__ fc2b,
                            const float* __restrict__ fcow, const float* __restrict__ fcob,
                            float* __restrict__ out)        // [256,1]
{
    int b = blockIdx.x * blockDim.x + threadIdx.x;
    if (b >= B_TOTAL) return;
    float h[32], u[32], v[16];
#pragma unroll
    for (int i = 0; i < 32; ++i) h[i] = h3[b * 32 + i];
#pragma unroll 4
    for (int o = 0; o < 32; ++o) {
        float s = fc1b[o];
#pragma unroll
        for (int i = 0; i < 32; ++i) s += fc1w[o * 32 + i] * h[i];
        u[o] = fmaxf(s, 0.0f);
    }
#pragma unroll 4
    for (int o = 0; o < 16; ++o) {
        float s = fc2b[o];
#pragma unroll
        for (int i = 0; i < 32; ++i) s += fc2w[o * 32 + i] * u[i];
        v[o] = fmaxf(s, 0.0f);
    }
    float s = fcob[0];
#pragma unroll
    for (int i = 0; i < 16; ++i) s += fcow[i] * v[i];
    out[b] = s;
}

extern "C" void kernel_launch(void* const* d_in, const int* in_sizes, int n_in,
                              void* d_out, int out_size, void* d_ws, size_t ws_size,
                              hipStream_t stream)
{
    // setup_inputs() order
    const float* x    = (const float*)d_in[0];
    const int*   lens = (const int*)  d_in[1];
    const float* Wih1 = (const float*)d_in[2];
    const float* Whh1 = (const float*)d_in[3];
    const float* bih1 = (const float*)d_in[4];
    const float* bhh1 = (const float*)d_in[5];
    const float* Wih2 = (const float*)d_in[6];
    const float* Whh2 = (const float*)d_in[7];
    const float* bih2 = (const float*)d_in[8];
    const float* bhh2 = (const float*)d_in[9];
    const float* Wih3 = (const float*)d_in[10];
    const float* Whh3 = (const float*)d_in[11];
    const float* bih3 = (const float*)d_in[12];
    const float* bhh3 = (const float*)d_in[13];
    const float* fc1w = (const float*)d_in[14];
    const float* fc1b = (const float*)d_in[15];
    const float* fc2w = (const float*)d_in[16];
    const float* fc2b = (const float*)d_in[17];
    const float* fcow = (const float*)d_in[18];
    const float* fcob = (const float*)d_in[19];

    // workspace: y1 (bf16 [B,T,128]) | y2 (bf16 [B,T,64]) | h3 (f32 [B,32])
    char* ws = (char*)d_ws;
    __bf16* y1 = (__bf16*)ws;
    size_t y1_bytes = (size_t)B_TOTAL * T_LEN * 128 * sizeof(__bf16);
    __bf16* y2 = (__bf16*)(ws + y1_bytes);
    size_t y2_bytes = (size_t)B_TOTAL * T_LEN * 64 * sizeof(__bf16);
    float* h3 = (float*)(ws + y1_bytes + y2_bytes);

    // dynamic LDS: swizzled weights (NT*KS*512 bf16) + A staging (16*K bf16)
    size_t smem1 = (size_t)(32 * 6) * 512 * 2 + 16 * 192 * 2;  // ~203 KB
    size_t smem2 = (size_t)(16 * 6) * 512 * 2 + 16 * 192 * 2;  // ~102 KB
    size_t smem3 = (size_t)(8  * 3) * 512 * 2 + 16 *  96 * 2;  // ~ 27 KB

    // layer 1: IN=64, H=128 -> 8 waves/block, 16 batch blocks
    lstm_layer_kernel<64, 128, true, true, false>
        <<<B_TOTAL / 16, 256, smem1, stream>>>(
            x, nullptr, lens, Wih1, Whh1, bih1, bhh1, y1, nullptr);

    // layer 2: IN=128, H=64 -> 4 waves/block
    lstm_layer_kernel<128, 64, false, true, false>
        <<<B_TOTAL / 16, 128, smem2, stream>>>(
            nullptr, y1, lens, Wih2, Whh2, bih2, bhh2, y2, nullptr);

    // layer 3: IN=64, H=32 -> 2 waves/block; only final h needed
    lstm_layer_kernel<64, 32, false, false, true>
        <<<B_TOTAL / 16, 64, smem3, stream>>>(
            nullptr, y2, lens, Wih3, Whh3, bih3, bhh3, nullptr, h3);

    // MLP head
    head_kernel<<<1, 256, 0, stream>>>(h3, fc1w, fc1b, fc2w, fc2b, fcow, fcob,
                                       (float*)d_out);
}
